// AttentionBlock_13958643712029
// MI455X (gfx1250) — matrix operand order repaired and checked by reading, not compiled, not run on hardware
//
#include <hip/hip_runtime.h>

// ---------------------------------------------------------------------------
// AttentionBlock for MI455X (gfx1250, wave32, WMMA).
// bf16 WMMA GEMMs with f32 accumulation. Fragments are loaded and consumed
// in the same iteration (no loop-carried PHIs -> no register copies); load/
// WMMA overlap comes from partial unrolling. 32x64 wave tiles in the big
// GEMMs for B-fragment reuse. Async global->LDS staging for the Q tile.
// ---------------------------------------------------------------------------

typedef __bf16 bf16;
typedef __attribute__((ext_vector_type(16))) __bf16 v16bf;
typedef __attribute__((ext_vector_type(8)))  __bf16 v8bf;
typedef __attribute__((ext_vector_type(4)))  __bf16 v4bf;
typedef __attribute__((ext_vector_type(8)))  float  v8f;
typedef __attribute__((ext_vector_type(4)))  float  v4f;
typedef __attribute__((ext_vector_type(4)))  int    v4i;

#define AS1 __attribute__((address_space(1)))
#define AS3 __attribute__((address_space(3)))

#if defined(__has_builtin)
#if __has_builtin(__builtin_amdgcn_global_load_async_to_lds_b128)
#define HAVE_ASYNC_LDS 1
#endif
#endif

constexpr int B = 8;
constexpr int S = 2048;
constexpr int E = 1024;
constexpr int D = 1024;
constexpr int M = B * S;           // 16384 total rows

// ---------------------------------------------------------------------------
// WMMA fragment helpers (contiguous k = 16h + s in both A and B, wave32)
// ---------------------------------------------------------------------------

// A-matrix 16x32 bf16, row-major source with row stride `ld` elements.
__device__ __forceinline__ v16bf load_a_frag(const bf16* base, int ld) {
  const int lane = threadIdx.x & 31;
  const int h = lane >> 4;
  const int m = lane & 15;
  return *(const v16bf*)(base + (size_t)m * ld + h * 16);
}

// B-matrix 32x16 bf16 from an N-major source (row = N, contiguous in K).
__device__ __forceinline__ v16bf load_b_frag(const bf16* base, int ld) {
  const int lane = threadIdx.x & 31;
  const int h = lane >> 4;
  const int n = lane & 15;
  return *(const v16bf*)(base + (size_t)n * ld + h * 16);
}

__device__ __forceinline__ v8f wmma_bf16(v16bf a, v16bf b, v8f c) {
  return __builtin_amdgcn_wmma_f32_16x16x32_bf16(false, a, false, b,
                                                 (short)0, c, false, false);
}

__device__ __forceinline__ void wait_async_lds() {
#if defined(__has_builtin)
#if __has_builtin(__builtin_amdgcn_s_wait_asynccnt)
  __builtin_amdgcn_s_wait_asynccnt(0);
  return;
#else
  asm volatile("s_wait_asynccnt 0x0" ::: "memory");
  return;
#endif
#endif
}

// ---------------------------------------------------------------------------
// Kernel 1: cast x (f32) -> bf16, vectorized x4
// ---------------------------------------------------------------------------
__global__ void cvt_x_kernel(const float* __restrict__ x,
                             bf16* __restrict__ xb, int n4) {
  int i = blockIdx.x * blockDim.x + threadIdx.x;
  if (i >= n4) return;
  v4f v = ((const v4f*)x)[i];
  v4bf o;
#pragma unroll
  for (int j = 0; j < 4; ++j) o[j] = (bf16)v[j];
  ((v4bf*)xb)[i] = o;
}

// ---------------------------------------------------------------------------
// Kernel 2: cast + transpose W [E,D] f32 -> Wt [D,E] bf16 (blockIdx.z picks W)
// ---------------------------------------------------------------------------
__global__ void cvt_wt_kernel(const float* __restrict__ Wq,
                              const float* __restrict__ Wk,
                              const float* __restrict__ Wv,
                              bf16* __restrict__ wqT,
                              bf16* __restrict__ wkT,
                              bf16* __restrict__ wvT) {
  const float* src = blockIdx.z == 0 ? Wq : (blockIdx.z == 1 ? Wk : Wv);
  bf16* dst = blockIdx.z == 0 ? wqT : (blockIdx.z == 1 ? wkT : wvT);
  int idx = blockIdx.x * blockDim.x + threadIdx.x;
  if (idx >= E * D) return;
  int k = idx / D;           // coalesced read of src
  int n = idx - k * D;
  dst[(size_t)n * E + k] = (bf16)src[idx];
}

// ---------------------------------------------------------------------------
// Kernel 3: QKV projections. Each wave: 32x64 tile (2 M x 4 N WMMA tiles,
// 6 fragment loads per 8 WMMAs). Q,K row-major [M,D]; V transposed [B,D,S].
// ---------------------------------------------------------------------------
__global__ void __launch_bounds__(128)
qkv_gemm_kernel(const bf16* __restrict__ X,
                const bf16* __restrict__ wqT, const bf16* __restrict__ wkT,
                const bf16* __restrict__ wvT,
                bf16* __restrict__ Qb, bf16* __restrict__ Kb,
                bf16* __restrict__ VT) {
  const int wave = threadIdx.x >> 5;
  const int lane = threadIdx.x & 31;
  const int m0 = blockIdx.x * 32;
  const int n0 = blockIdx.y * 256 + wave * 64;
  const int which = blockIdx.z;
  const bf16* W = which == 0 ? wqT : (which == 1 ? wkT : wvT);

  const bf16* arow0 = X + (size_t)m0 * E;
  const bf16* arow1 = arow0 + (size_t)16 * E;
  v8f acc0[4] = {};
  v8f acc1[4] = {};

#pragma unroll 4
  for (int k0 = 0; k0 < E; k0 += 32) {
    v16bf a0 = load_a_frag(arow0 + k0, E);
    v16bf a1 = load_a_frag(arow1 + k0, E);
#pragma unroll
    for (int j = 0; j < 4; ++j) {
      v16bf bj = load_b_frag(W + (size_t)(n0 + j * 16) * E + k0, E);
      acc0[j] = wmma_bf16(a0, bj, acc0[j]);
      acc1[j] = wmma_bf16(a1, bj, acc1[j]);
    }
  }

  const int h = lane >> 4, nlo = lane & 15;
#pragma unroll
  for (int mi = 0; mi < 2; ++mi) {
#pragma unroll
    for (int j = 0; j < 4; ++j) {
#pragma unroll
      for (int r = 0; r < 8; ++r) {
        int row = m0 + mi * 16 + r + 8 * h;
        int col = n0 + j * 16 + nlo;
        bf16 val = (bf16)(mi == 0 ? acc0[j][r] : acc1[j][r]);
        if (which == 0) {
          Qb[(size_t)row * D + col] = val;
        } else if (which == 1) {
          Kb[(size_t)row * D + col] = val;
        } else {
          int bb = row >> 11;            // row / S
          int ss = row & (S - 1);
          VT[((size_t)bb * D + col) * S + ss] = val;
        }
      }
    }
  }
}

// ---------------------------------------------------------------------------
// Kernel 4: scores. 4 waves per block share one async-LDS-staged Q tile;
// wave w covers k-columns [w*512, (w+1)*512). Emits unnormalized
// p = mask * exp(s/sqrt(D)) in bf16 plus per-wave partial row sums
// (deterministic: no float atomics). The reference's renormalized masked
// softmax reduces to m*exp(s)/(sum(m*exp(s))+eps).
// ---------------------------------------------------------------------------
__global__ void __launch_bounds__(128)
scores_kernel(const bf16* __restrict__ Qb, const bf16* __restrict__ Kb,
              const int* __restrict__ mask,
              bf16* __restrict__ P, float* __restrict__ rowsum_part) {
  __shared__ bf16 qs[16 * D];          // 32 KB
  const int wave = threadIdx.x >> 5;
  const int lane = threadIdx.x & 31;
  const int b = blockIdx.x / (S / 16);
  const int q0 = (blockIdx.x % (S / 16)) * 16;

  // stage Q tile (16 x D bf16 = 32 KB) into LDS via async DMA
  {
    const char* src = (const char*)(Qb + ((size_t)b * S + q0) * D);
#if defined(HAVE_ASYNC_LDS)
    AS3 char* dst = (AS3 char*)(char*)qs;
    for (int i = threadIdx.x; i < (16 * D * 2) / 16; i += 128) {
      __builtin_amdgcn_global_load_async_to_lds_b128(
          (AS1 v4i*)(src + (size_t)i * 16),
          (AS3 v4i*)(dst + i * 16), 0, 0);
    }
    wait_async_lds();
#else
    const v8bf* s8 = (const v8bf*)src;
    v8bf* d8 = (v8bf*)qs;
    for (int i = threadIdx.x; i < 16 * D / 8; i += 128) d8[i] = s8[i];
#endif
  }
  __syncthreads();

  const int h = lane >> 4, nlo = lane & 15;
  const float scale = 0.03125f;        // 1/sqrt(D)
  const int kw0 = wave * (S / 4);      // this wave's k-range start
  float rs[8] = {};

  for (int k0 = kw0; k0 < kw0 + S / 4; k0 += 32) {
    const bf16* kb0 = Kb + ((size_t)b * S + k0) * D;
    v8f acc0 = {}, acc1 = {};

#pragma unroll 4
    for (int d = 0; d < D; d += 32) {
      v16bf a  = load_a_frag(qs + d, D);
      v16bf b0 = load_b_frag(kb0 + d, D);
      v16bf b1 = load_b_frag(kb0 + (size_t)16 * D + d, D);
      acc0 = wmma_bf16(a, b0, acc0);
      acc1 = wmma_bf16(a, b1, acc1);
    }

    const int* mrow = mask + ((size_t)b * S + q0) * S + k0;
    bf16* prow = P + ((size_t)b * S + q0) * S + k0;
#pragma unroll
    for (int r = 0; r < 8; ++r) {
      int row = r + 8 * h;
      size_t rbase = (size_t)row * S;
      int m0v = mrow[rbase + nlo];
      int m1v = mrow[rbase + 16 + nlo];
      float p0 = m0v ? __expf(acc0[r] * scale) : 0.0f;
      float p1 = m1v ? __expf(acc1[r] * scale) : 0.0f;
      rs[r] += p0 + p1;
      prow[rbase + nlo] = (bf16)p0;
      prow[rbase + 16 + nlo] = (bf16)p1;
    }
  }

  // reduce each row's partial sum across its 16-lane half-wave
#pragma unroll
  for (int r = 0; r < 8; ++r) {
    float v = rs[r];
    v += __shfl_xor(v, 1, 32);
    v += __shfl_xor(v, 2, 32);
    v += __shfl_xor(v, 4, 32);
    v += __shfl_xor(v, 8, 32);
    if (nlo == 0)
      rowsum_part[(size_t)(b * S + q0 + r + 8 * h) * 4 + wave] = v;
  }
}

// ---------------------------------------------------------------------------
// Kernel 5: out = (P @ V) / (rowsum + eps). Each wave: 32x64 tile.
// ---------------------------------------------------------------------------
__global__ void __launch_bounds__(128)
out_gemm_kernel(const bf16* __restrict__ P, const bf16* __restrict__ VT,
                const float* __restrict__ rowsum_part, float* __restrict__ out) {
  const int wave = threadIdx.x >> 5;
  const int lane = threadIdx.x & 31;
  const int b = blockIdx.x / (S / 32);
  const int q0 = (blockIdx.x % (S / 32)) * 32;
  const int n0 = blockIdx.y * 256 + wave * 64;

  const bf16* prow0 = P + ((size_t)b * S + q0) * S;
  const bf16* prow1 = prow0 + (size_t)16 * S;
  const bf16* vtb = VT + (size_t)b * D * S;
  v8f acc0[4] = {};
  v8f acc1[4] = {};

#pragma unroll 4
  for (int k0 = 0; k0 < S; k0 += 32) {
    v16bf a0 = load_a_frag(prow0 + k0, S);
    v16bf a1 = load_a_frag(prow1 + k0, S);
#pragma unroll
    for (int j = 0; j < 4; ++j) {
      v16bf bj = load_b_frag(vtb + (size_t)(n0 + j * 16) * S + k0, S);
      acc0[j] = wmma_bf16(a0, bj, acc0[j]);
      acc1[j] = wmma_bf16(a1, bj, acc1[j]);
    }
  }

  const int h = lane >> 4, nlo = lane & 15;
#pragma unroll
  for (int mi = 0; mi < 2; ++mi) {
#pragma unroll
    for (int r = 0; r < 8; ++r) {
      int row = q0 + mi * 16 + r + 8 * h;
      const float* rp = rowsum_part + (size_t)(b * S + row) * 4;
      float inv = 1.0f / (rp[0] + rp[1] + rp[2] + rp[3] + 1e-20f);
#pragma unroll
      for (int j = 0; j < 4; ++j) {
        float v = (mi == 0 ? acc0[j][r] : acc1[j][r]) * inv;
        out[((size_t)b * S + row) * D + n0 + j * 16 + nlo] = v;
      }
    }
  }
}

// ---------------------------------------------------------------------------
// Host-side launch
// ---------------------------------------------------------------------------
extern "C" void kernel_launch(void* const* d_in, const int* in_sizes, int n_in,
                              void* d_out, int out_size, void* d_ws, size_t ws_size,
                              hipStream_t stream) {
  const float* x    = (const float*)d_in[0];
  const int*   mask = (const int*)d_in[1];
  const float* Wq   = (const float*)d_in[2];
  const float* Wk   = (const float*)d_in[3];
  const float* Wv   = (const float*)d_in[4];
  float* out = (float*)d_out;

  // workspace carve-up (~198 MiB total)
  bf16* xb  = (bf16*)d_ws;                    // [M,E]
  bf16* qb  = xb  + (size_t)M * E;            // [M,D]
  bf16* kb  = qb  + (size_t)M * D;            // [M,D]
  bf16* vt  = kb  + (size_t)M * D;            // [B,D,S]
  bf16* wqT = vt  + (size_t)M * D;            // [D,E]
  bf16* wkT = wqT + (size_t)E * D;
  bf16* wvT = wkT + (size_t)E * D;
  bf16* Pm  = wvT + (size_t)E * D;            // [B,S,S]
  float* rowsum_part = (float*)(Pm + (size_t)B * S * S);   // [B,S,4]

  {
    int n4 = M * E / 4;
    cvt_x_kernel<<<(n4 + 255) / 256, 256, 0, stream>>>(x, xb, n4);
  }
  {
    dim3 g((E * D + 255) / 256, 1, 3);
    cvt_wt_kernel<<<g, 256, 0, stream>>>(Wq, Wk, Wv, wqT, wkT, wvT);
  }
  {
    dim3 g(M / 32, D / 256, 3);
    qkv_gemm_kernel<<<g, 128, 0, stream>>>(xb, wqT, wkT, wvT, qb, kb, vt);
  }
  {
    scores_kernel<<<B * S / 16, 128, 0, stream>>>(qb, kb, mask, Pm, rowsum_part);
  }
  {
    dim3 g(B * S / 32, D / 256);
    out_gemm_kernel<<<g, 128, 0, stream>>>(Pm, vt, rowsum_part, out);
  }
}
